// GCN_49082886259351
// MI455X (gfx1250) — matrix operand order repaired and checked by reading, not compiled, not run on hardware
//
#include <hip/hip_runtime.h>
#include <hip/hip_bf16.h>

typedef __bf16 v16bf __attribute__((ext_vector_type(16)));
typedef float  v8f   __attribute__((ext_vector_type(8)));

#define EPS_BN 1e-5f

// ---------------- degree / dinv ----------------
__global__ void k_init_deg(float* __restrict__ deg, int n) {
  int i = blockIdx.x * blockDim.x + threadIdx.x;
  if (i < n) deg[i] = 1.0f;  // self-loop weight
}

__global__ void k_deg_accum(float* __restrict__ deg, const int* __restrict__ col,
                            const float* __restrict__ w, int E) {
  int e = blockIdx.x * blockDim.x + threadIdx.x;
  if (e < E) unsafeAtomicAdd(&deg[col[e]], w[e]);
}

__global__ void k_deg_rsqrt(float* __restrict__ deg, int n) {
  int i = blockIdx.x * blockDim.x + threadIdx.x;
  if (i < n) {
    float d = deg[i];
    deg[i] = (d > 0.0f) ? rsqrtf(d) : 0.0f;
  }
}

// ---------------- pack W into WMMA-B lane layout (bf16 hi/lo) ----------------
// flat index = ((ktile*NT + ntile)*32 + lane)*16 + i
// lane<16 : N = ntile*16+lane,      K = ktile*32 + (i&7) + (i>>3)*16
// lane>=16: N = ntile*16+(lane-16), K = ktile*32 + 8 + (i&7) + (i>>3)*16
__global__ void k_pack_w(const float* __restrict__ W, __bf16* __restrict__ Bhi,
                         __bf16* __restrict__ Blo, int dout, int total) {
  int tid = blockIdx.x * blockDim.x + threadIdx.x;
  if (tid >= total) return;
  int i    = tid & 15;
  int lane = (tid >> 4) & 31;
  int t    = tid >> 9;
  int NT    = dout >> 4;
  int ntile = t % NT;
  int ktile = t / NT;
  int k    = ktile * 32 + (i & 7) + ((i >> 3) << 4) + ((lane >= 16) ? 8 : 0);
  int ncol = ntile * 16 + (lane & 15);
  float w  = W[k * dout + ncol];
  __bf16 h = (__bf16)w;
  Bhi[tid] = h;
  Blo[tid] = (__bf16)(w - (float)h);
}

// ---- WMMA GEMM with (optional) fused self-loop epilogue ----
// XW[N x DOUT]  = A[N x K] @ W             (split-bf16, f32 accumulate)
// if Agg: Agg[N x DOUT] = XW * dinv[row]^2 (self-loop init for scatter-add)
// NOTE: Agg must NOT alias A (cross-block race); pass nullptr to skip.
template <int DOUT>
__global__ void k_gemm(const float* __restrict__ A, const __bf16* __restrict__ Bhi,
                       const __bf16* __restrict__ Blo, const float* __restrict__ dinv,
                       float* __restrict__ XW, float* __restrict__ Agg,
                       int nRows, int K) {
  constexpr int NT = DOUT / 16;
  const int lane  = threadIdx.x & 31;
  const int ntile = threadIdx.x >> 5;           // one wave per 16x16 C tile
  const int hi16  = lane >> 4;                  // 0 for lanes 0-15, 1 for 16-31
  int row = blockIdx.x * 16 + (lane & 15);
  if (row >= nRows) row = nRows - 1;            // clamp (keeps EXEC full for WMMA)
  const int koff = hi16 ? 8 : 0;                // A-layout K offset for upper lanes

  v8f acc = {};
  const int ktiles = K >> 5;
  for (int kt = 0; kt < ktiles; ++kt) {
    // A 16x32 bf16 layout: elems 0..7 -> K = base..base+7, elems 8..15 -> K = base+16..base+23
    const float* ap = A + (size_t)row * K + kt * 32 + koff;
    float4 p0 = *(const float4*)(ap);
    float4 p1 = *(const float4*)(ap + 4);
    float4 p2 = *(const float4*)(ap + 16);
    float4 p3 = *(const float4*)(ap + 20);
    float av[16] = {p0.x, p0.y, p0.z, p0.w, p1.x, p1.y, p1.z, p1.w,
                    p2.x, p2.y, p2.z, p2.w, p3.x, p3.y, p3.z, p3.w};
    v16bf ahi, alo;
#pragma unroll
    for (int i = 0; i < 16; ++i) {
      __bf16 h = (__bf16)av[i];
      ahi[i] = h;
      alo[i] = (__bf16)(av[i] - (float)h);
    }
    const size_t bidx = (((size_t)kt * NT + ntile) * 32 + lane) * 16;
    v16bf bhi = *(const v16bf*)(Bhi + bidx);
    v16bf blo = *(const v16bf*)(Blo + bidx);
    acc = __builtin_amdgcn_wmma_f32_16x16x32_bf16(false, ahi, false, bhi, (short)0, acc, false, false);
    acc = __builtin_amdgcn_wmma_f32_16x16x32_bf16(false, ahi, false, blo, (short)0, acc, false, false);
    acc = __builtin_amdgcn_wmma_f32_16x16x32_bf16(false, alo, false, bhi, (short)0, acc, false, false);
  }

  // C/D layout: VGPR r -> M = r (lanes 0-15) / r+8 (lanes 16-31); N = lane&15
  const int crow0 = blockIdx.x * 16 + (hi16 ? 8 : 0);
  const int ccol  = ntile * 16 + (lane & 15);
  const bool fullTile = (blockIdx.x * 16 + 16 <= (unsigned)nRows);
  if (Agg != nullptr) {
    if (fullTile) {
#pragma unroll
      for (int r = 0; r < 8; ++r) {
        const int crow = crow0 + r;
        const size_t o = (size_t)crow * DOUT + ccol;
        const float v  = acc[r];
        const float dv = dinv[crow];
        XW[o]  = v;
        Agg[o] = v * dv * dv;
      }
    } else {
#pragma unroll
      for (int r = 0; r < 8; ++r) {
        const int crow = crow0 + r;
        if (crow < nRows) {
          const size_t o = (size_t)crow * DOUT + ccol;
          const float v  = acc[r];
          const float dv = dinv[crow];
          XW[o]  = v;
          Agg[o] = v * dv * dv;
        }
      }
    }
  } else {
    if (fullTile) {
#pragma unroll
      for (int r = 0; r < 8; ++r)
        XW[(size_t)(crow0 + r) * DOUT + ccol] = acc[r];
    } else {
#pragma unroll
      for (int r = 0; r < 8; ++r) {
        const int crow = crow0 + r;
        if (crow < nRows) XW[(size_t)crow * DOUT + ccol] = acc[r];
      }
    }
  }
}

// ---------------- self-loop contribution (fallback path): out = xw * dinv[i]^2 ----
__global__ void k_selfloop(const float* __restrict__ xw, const float* __restrict__ dinv,
                           float* __restrict__ out, int total, int shift) {
  int idx = blockIdx.x * blockDim.x + threadIdx.x;
  if (idx >= total) return;
  float dv = dinv[idx >> shift];
  out[idx] = xw[idx] * dv * dv;
}

// ---------------- wave-per-edge gather + atomic scatter ----------------
template <int D>
__global__ void k_edge_scatter(const float* __restrict__ xw, const int* __restrict__ row,
                               const int* __restrict__ col, const float* __restrict__ w,
                               const float* __restrict__ dinv, float* __restrict__ out, int E) {
  int e = (blockIdx.x * blockDim.x + threadIdx.x) >> 5;
  int lane = threadIdx.x & 31;
  if (e >= E) return;
  int r = row[e], c = col[e];
  float nrm = dinv[r] * w[e] * dinv[c];
  constexpr int PER = D / 32;
  const float* src = xw + (size_t)r * D + lane * PER;
  float* dst = out + (size_t)c * D + lane * PER;
  if constexpr (PER == 4) {
    float4 v = *(const float4*)src;
    unsafeAtomicAdd(dst + 0, v.x * nrm);
    unsafeAtomicAdd(dst + 1, v.y * nrm);
    unsafeAtomicAdd(dst + 2, v.z * nrm);
    unsafeAtomicAdd(dst + 3, v.w * nrm);
  } else {
    float2 v = *(const float2*)src;
    unsafeAtomicAdd(dst + 0, v.x * nrm);
    unsafeAtomicAdd(dst + 1, v.y * nrm);
  }
}

// ---------------- bias + eval-mode BN + ReLU (in place, D=128) ----------------
__global__ void k_bias_bn_relu(float* __restrict__ h, const float* __restrict__ b,
                               const float* __restrict__ g, const float* __restrict__ be,
                               const float* __restrict__ rm, const float* __restrict__ rv,
                               int total) {
  int idx = blockIdx.x * blockDim.x + threadIdx.x;
  if (idx >= total) return;
  int j = idx & 127;
  float v = h[idx] + b[j];
  float y = (v - rm[j]) * rsqrtf(rv[j] + EPS_BN) * g[j] + be[j];
  h[idx] = fmaxf(y, 0.0f);
}

// ---------------- bias + row L2 normalize (D=64, wave per node) ----------------
__global__ void k_bias_l2norm(const float* __restrict__ agg, const float* __restrict__ b3,
                              float* __restrict__ out, int n) {
  int node = (blockIdx.x * blockDim.x + threadIdx.x) >> 5;
  int lane = threadIdx.x & 31;
  if (node >= n) return;
  float2 v  = *(const float2*)(agg + (size_t)node * 64 + lane * 2);
  float2 bb = *(const float2*)(b3 + lane * 2);
  float h0 = v.x + bb.x, h1 = v.y + bb.y;
  float ss = h0 * h0 + h1 * h1;
#pragma unroll
  for (int off = 16; off > 0; off >>= 1) ss += __shfl_xor(ss, off, 32);
  float scale = 1.0f / fmaxf(sqrtf(ss), 1e-12f);
  float2 o;
  o.x = h0 * scale;
  o.y = h1 * scale;
  *(float2*)(out + (size_t)node * 64 + lane * 2) = o;
}

extern "C" void kernel_launch(void* const* d_in, const int* in_sizes, int n_in,
                              void* d_out, int out_size, void* d_ws, size_t ws_size,
                              hipStream_t stream) {
  const float* x   = (const float*)d_in[0];
  const int*   ei  = (const int*)d_in[1];
  const float* ew  = (const float*)d_in[2];
  const float* W1  = (const float*)d_in[3];
  const float* b1  = (const float*)d_in[4];
  const float* g1  = (const float*)d_in[5];
  const float* be1 = (const float*)d_in[6];
  const float* rm1 = (const float*)d_in[7];
  const float* rv1 = (const float*)d_in[8];
  const float* W2  = (const float*)d_in[9];
  const float* b2  = (const float*)d_in[10];
  const float* g2  = (const float*)d_in[11];
  const float* be2 = (const float*)d_in[12];
  const float* rm2 = (const float*)d_in[13];
  const float* rv2 = (const float*)d_in[14];
  const float* W3  = (const float*)d_in[15];
  const float* b3  = (const float*)d_in[16];

  const int N = in_sizes[0] / 128;
  const int E = in_sizes[1] / 2;
  const int* row = ei;
  const int* col = ei + E;
  float* out = (float*)d_out;

  // workspace carve-out
  char* ws = (char*)d_ws;
  size_t off = 0;
  auto carve = [&](size_t bytes) -> void* {
    void* p = ws + off;
    off += (bytes + 255) & ~(size_t)255;
    return p;
  };
  const size_t hbytes = (size_t)N * 128 * 4;
  float*  dinv = (float*)carve((size_t)N * 4);
  float*  bufA = (float*)carve(hbytes);   // XW
  float*  bufB = (float*)carve(hbytes);   // Agg / H
  __bf16* W1hi = (__bf16*)carve(128 * 128 * 2);
  __bf16* W1lo = (__bf16*)carve(128 * 128 * 2);
  __bf16* W2hi = (__bf16*)carve(128 * 128 * 2);
  __bf16* W2lo = (__bf16*)carve(128 * 128 * 2);
  __bf16* W3hi = (__bf16*)carve(128 * 64 * 2);
  __bf16* W3lo = (__bf16*)carve(128 * 64 * 2);
  // third H buffer only if workspace is large enough (decision is a pure
  // function of ws_size -> deterministic across calls)
  const bool has3 = (off + hbytes) <= ws_size;
  float* bufC = has3 ? (float*)carve(hbytes) : nullptr;

  const int TB = 256;
  const int nBlk  = (N + TB - 1) / TB;
  const int eBlk  = (E + TB - 1) / TB;
  const int neBlk = ((size_t)N * 128 + TB - 1) / TB;
  const int n3Blk = ((size_t)N * 64 + TB - 1) / TB;
  const int waveE = (E + 7) / 8;     // wave-per-edge, 8 waves/block
  const int waveN = (N + 7) / 8;     // wave-per-node
  const int mTiles = (N + 15) / 16;

  // degree -> dinv
  k_init_deg<<<nBlk, TB, 0, stream>>>(dinv, N);
  k_deg_accum<<<eBlk, TB, 0, stream>>>(dinv, col, ew, E);
  k_deg_rsqrt<<<nBlk, TB, 0, stream>>>(dinv, N);

  // pack weights into WMMA B layout (hi/lo bf16)
  k_pack_w<<<(128 * 128 + TB - 1) / TB, TB, 0, stream>>>(W1, W1hi, W1lo, 128, 128 * 128);
  k_pack_w<<<(128 * 128 + TB - 1) / TB, TB, 0, stream>>>(W2, W2hi, W2lo, 128, 128 * 128);
  k_pack_w<<<(128 * 64  + TB - 1) / TB, TB, 0, stream>>>(W3, W3hi, W3lo, 64, 128 * 64);

  if (has3) {
    // fully fused flow: Agg never aliases the GEMM's A operand
    // ---- layer 1: A=x, XW=bufA, Agg=bufB ----
    k_gemm<128><<<mTiles, 256, 0, stream>>>(x, W1hi, W1lo, dinv, bufA, bufB, N, 128);
    k_edge_scatter<128><<<waveE, 256, 0, stream>>>(bufA, row, col, ew, dinv, bufB, E);
    k_bias_bn_relu<<<neBlk, TB, 0, stream>>>(bufB, b1, g1, be1, rm1, rv1, N * 128);
    // ---- layer 2: A=bufB, XW=bufA, Agg=bufC ----
    k_gemm<128><<<mTiles, 256, 0, stream>>>(bufB, W2hi, W2lo, dinv, bufA, bufC, N, 128);
    k_edge_scatter<128><<<waveE, 256, 0, stream>>>(bufA, row, col, ew, dinv, bufC, E);
    k_bias_bn_relu<<<neBlk, TB, 0, stream>>>(bufC, b2, g2, be2, rm2, rv2, N * 128);
    // ---- layer 3: A=bufC, XW=bufA, Agg=bufB ----
    k_gemm<64><<<mTiles, 128, 0, stream>>>(bufC, W3hi, W3lo, dinv, bufA, bufB, N, 128);
    k_edge_scatter<64><<<waveE, 256, 0, stream>>>(bufA, row, col, ew, dinv, bufB, E);
    k_bias_l2norm<<<waveN, 256, 0, stream>>>(bufB, b3, out, N);
  } else {
    // 2-buffer fallback: fused only for layer 1 (A=x is an input, no alias);
    // layers 2/3 do the self-loop init in a separate kernel after the GEMM.
    // ---- layer 1 ----
    k_gemm<128><<<mTiles, 256, 0, stream>>>(x, W1hi, W1lo, dinv, bufA, bufB, N, 128);
    k_edge_scatter<128><<<waveE, 256, 0, stream>>>(bufA, row, col, ew, dinv, bufB, E);
    k_bias_bn_relu<<<neBlk, TB, 0, stream>>>(bufB, b1, g1, be1, rm1, rv1, N * 128);
    // ---- layer 2 ----
    k_gemm<128><<<mTiles, 256, 0, stream>>>(bufB, W2hi, W2lo, dinv, bufA, nullptr, N, 128);
    k_selfloop<<<neBlk, TB, 0, stream>>>(bufA, dinv, bufB, N * 128, 7);
    k_edge_scatter<128><<<waveE, 256, 0, stream>>>(bufA, row, col, ew, dinv, bufB, E);
    k_bias_bn_relu<<<neBlk, TB, 0, stream>>>(bufB, b2, g2, be2, rm2, rv2, N * 128);
    // ---- layer 3 ----
    k_gemm<64><<<mTiles, 128, 0, stream>>>(bufB, W3hi, W3lo, dinv, bufA, nullptr, N, 128);
    k_selfloop<<<n3Blk, TB, 0, stream>>>(bufA, dinv, bufB, N * 64, 6);
    k_edge_scatter<64><<<waveE, 256, 0, stream>>>(bufA, row, col, ew, dinv, bufB, E);
    k_bias_l2norm<<<waveN, 256, 0, stream>>>(bufB, b3, out, N);
  }
}
